// GNN_auto_56942676411150
// MI455X (gfx1250) — matrix operand order repaired and checked by reading, not compiled, not run on hardware
//
#include <hip/hip_runtime.h>
#include <math.h>

// ---------------------------------------------------------------------------
// CDNA5 (gfx1250) GNN forward. wave32 everywhere.
// GEMMs: v_wmma_f32_16x16x32_f16 on pre-converted f16 operands (f32 accum).
// Edge phase: one wave/edge, float4 lanes, butterfly shfl, f32 global atomics,
// global_prefetch_b8 warm-up for the message-phase rows.
// ---------------------------------------------------------------------------

typedef __attribute__((ext_vector_type(16))) _Float16 v16h;
typedef __attribute__((ext_vector_type(8)))  float    v8f;

#define HD 128          // hidden dim == attn dim
#define GN 384          // 3*HD GRU gate dim

__device__ __forceinline__ float sigmoidf_(float x) { return 1.f / (1.f + __expf(-x)); }

// ---------------------------------------------------------------------------
// f16 tile loads. k order used by this code (wave32; contiguous, not the ISA table order):
//   A 16x32 f16: lane = row%16, half = lane/16; K elems = kBase+16h .. +15
//                -> one contiguous 32B run per lane (matches B).
//   B 32x16 f16 (B[k][n] = W[n][k]):  lane = col%16, half = lane/16;
//                K elems = kBase+16h .. +15 -> one contiguous 32B run per lane.
// N must be a multiple of 16 at every call site (HD=128, GN=384) -> no B guard.
// ---------------------------------------------------------------------------
__device__ __forceinline__ v16h load_a16_fast(const _Float16* __restrict__ A, int K,
                                              int r, int kBase) {
  int half = (threadIdx.x & 31) >> 4;
  const _Float16* row = A + (size_t)r * K + kBase + 16 * half;
  v16h out;
#pragma unroll
  for (int j = 0; j < 8; ++j) out[j] = row[j];
#pragma unroll
  for (int j = 0; j < 8; ++j) out[8 + j] = row[8 + j];
  return out;
}

__device__ __forceinline__ v16h load_b16(const _Float16* __restrict__ W, int K,
                                         int nBase, int kBase) {
  int lane = threadIdx.x & 31;
  const _Float16* row = W + (size_t)(nBase + (lane & 15)) * K + kBase + 16 * (lane >> 4);
  v16h out;
#pragma unroll
  for (int t = 0; t < 16; ++t) out[t] = row[t];
  return out;
}

// 16x16 f32 C tile = A[M x K] @ W[N x K]^T ; K multiple of 32.
// fullM selects the unguarded fast path (wave-uniform), used for all full
// 16-row tiles; the guarded path only runs for the one partial relation tile.
__device__ __forceinline__ v8f wmma_tile16(const _Float16* __restrict__ A,
                                           const _Float16* __restrict__ W,
                                           int M, int K, int mBase, int nBase, bool fullM) {
  v8f acc = {};
  int lane = threadIdx.x & 31;
  int r = mBase + (lane & 15);
  if (fullM) {
    for (int kBase = 0; kBase < K; kBase += 32) {
      v16h a = load_a16_fast(A, K, r, kBase);
      v16h b = load_b16(W, K, nBase, kBase);
      acc = __builtin_amdgcn_wmma_f32_16x16x32_f16(false, a, false, b, (short)0, acc,
                                                   false, false);
    }
  } else {
    bool in = (r < M);
    int rc = in ? r : 0;                      // clamp; masked below
    for (int kBase = 0; kBase < K; kBase += 32) {
      v16h a = load_a16_fast(A, K, rc, kBase);
      if (!in) {
#pragma unroll
        for (int t = 0; t < 16; ++t) a[t] = (_Float16)0.f;
      }
      v16h b = load_b16(W, K, nBase, kBase);
      acc = __builtin_amdgcn_wmma_f32_16x16x32_f16(false, a, false, b, (short)0, acc,
                                                   false, false);
    }
  }
  return acc;
}

// ---------------------------------------------------------------------------
// Generic C = act(A @ W^T + bias). 4 waves/block, one 16x16 tile per wave
// (waves of a block share the same A rows -> WGP-cache reuse).
// ---------------------------------------------------------------------------
__global__ void wmma_gemm_atb(const _Float16* __restrict__ A, const _Float16* __restrict__ W,
                              const float* __restrict__ bias, float* __restrict__ C,
                              int M, int N, int K, int applyRelu) {
  int mBase = blockIdx.x * 16;
  int nBase = (blockIdx.y * 4 + (threadIdx.x >> 5)) * 16;
  if (nBase >= N) return;                 // wave-uniform
  bool fullM = (mBase + 16 <= M);
  v8f acc = wmma_tile16(A, W, M, K, mBase, nBase, fullM);
  int lane = threadIdx.x & 31;
  int col = nBase + (lane & 15);
  int rowb = mBase + (lane >> 4) * 8;
  float bv = (bias != nullptr) ? bias[col] : 0.f;
#pragma unroll
  for (int v = 0; v < 8; ++v) {
    int row = rowb + v;
    if (row < M) {
      float val = acc[v] + bv;
      if (applyRelu) val = fmaxf(val, 0.f);
      C[(size_t)row * N + col] = val;
    }
  }
}

// ---------------------------------------------------------------------------
// Edge phase: one wave per edge, lane l owns components [4l, 4l+4).
// ---------------------------------------------------------------------------
__global__ void edge_kernel(const int* __restrict__ e_sub, const int* __restrict__ e_rel,
                            const int* __restrict__ e_batch, const int* __restrict__ e_obj,
                            const float* __restrict__ wsn, const float* __restrict__ wr,
                            const float* __restrict__ wqr, const float* __restrict__ hidden,
                            const float* __restrict__ hr_tab,
                            const float* __restrict__ walphaW, const float* __restrict__ walphaB,
                            float* __restrict__ agg, int nEdge) {
  int wid = blockIdx.x * (blockDim.x >> 5) + (threadIdx.x >> 5);
  int lane = threadIdx.x & 31;
  if (wid >= nEdge) return;
  int sub = e_sub[wid], rel = e_rel[wid], bq = e_batch[wid], obj = e_obj[wid];

  // Warm the message-phase rows while the attention gathers are in flight
  // (lowers to global_prefetch_b8 on gfx1250).
  __builtin_prefetch(hidden + (size_t)sub * HD + lane * 4, 0, 3);
  __builtin_prefetch(hr_tab + (size_t)rel * HD + lane * 4, 0, 3);
  __builtin_prefetch(agg    + (size_t)obj * HD + lane * 4, 1, 3);

  const float4 s4 = ((const float4*)(wsn + (size_t)sub * HD))[lane];
  const float4 r4 = ((const float4*)(wr  + (size_t)rel * HD))[lane];
  const float4 q4 = ((const float4*)(wqr + (size_t)bq  * HD))[lane];
  const float4 w4 = ((const float4*)walphaW)[lane];

  float ax = fmaxf(s4.x + r4.x + q4.x, 0.f);
  float ay = fmaxf(s4.y + r4.y + q4.y, 0.f);
  float az = fmaxf(s4.z + r4.z + q4.z, 0.f);
  float aw = fmaxf(s4.w + r4.w + q4.w, 0.f);
  float part = ax * w4.x + ay * w4.y + az * w4.z + aw * w4.w;
#pragma unroll
  for (int off = 16; off > 0; off >>= 1) part += __shfl_xor(part, off, 32);
  float alpha = sigmoidf_(part + walphaB[0]);

  const float4 h4 = ((const float4*)(hidden + (size_t)sub * HD))[lane];
  const float4 t4 = ((const float4*)(hr_tab + (size_t)rel * HD))[lane];
  float* dst = agg + (size_t)obj * HD + lane * 4;
  atomicAdd(dst + 0, alpha * h4.x * t4.x);
  atomicAdd(dst + 1, alpha * h4.y * t4.y);
  atomicAdd(dst + 2, alpha * h4.z * t4.z);
  atomicAdd(dst + 3, alpha * h4.w * t4.w);
}

// ---------------------------------------------------------------------------
// Fused GRU for a 16-node tile. 8 waves x 6 WMMA tile-GEMMs:
//   gi = h_layer @ Wih^T + bih (24 N-tiles), gh = h0 @ Whh^T + bhh (24)
// into LDS (49KB << 320KB/WGP), then gates + dead mask, in-place on hidden.
// ---------------------------------------------------------------------------
__global__ void gru_kernel(const _Float16* __restrict__ hlay16,   // gi A operand
                           const _Float16* __restrict__ hid16,    // gh A operand (h0)
                           const float* __restrict__ h_layer,     // f32, dead-mask source
                           float* __restrict__ hidden,            // f32 h0 in / h_new out
                           const _Float16* __restrict__ Wih16, const _Float16* __restrict__ Whh16,
                           const float* __restrict__ bih, const float* __restrict__ bhh,
                           int M) {
  __shared__ float gi[16][GN];
  __shared__ float gh[16][GN];
  __shared__ float psum[16][16];
  __shared__ float keep[16];
  int mBase = blockIdx.x * 16;
  int wave = threadIdx.x >> 5;
  int lane = threadIdx.x & 31;
  bool fullM = (mBase + 16 <= M);

  for (int t = wave * 6; t < wave * 6 + 6; ++t) {
    int isGH = (t >= 24);
    int ntile = isGH ? (t - 24) : t;
    const _Float16* A = isGH ? hid16 : hlay16;
    const _Float16* W = isGH ? Whh16 : Wih16;
    const float* b = isGH ? bhh : bih;
    v8f acc = wmma_tile16(A, W, M, HD, mBase, ntile * 16, fullM);
    int col = ntile * 16 + (lane & 15);
    int rowb = (lane >> 4) * 8;
    float bv = b[col];
#pragma unroll
    for (int v = 0; v < 8; ++v) {
      if (mBase + rowb + v < M) {
        if (isGH) gh[rowb + v][col] = acc[v] + bv;
        else      gi[rowb + v][col] = acc[v] + bv;
      }
    }
  }

  // dead-node mask from f32 h_layer (f16 could flush tiny positives to 0)
  {
    int r = threadIdx.x >> 4, c = threadIdx.x & 15;
    float s = 0.f;
    if (mBase + r < M) {
      const float* p = h_layer + (size_t)(mBase + r) * HD + c * 8;
#pragma unroll
      for (int j = 0; j < 8; ++j) s += p[j];
    }
    psum[r][c] = s;
  }
  __syncthreads();
  if (threadIdx.x < 16) {
    float s = 0.f;
#pragma unroll
    for (int c = 0; c < 16; ++c) s += psum[threadIdx.x][c];
    keep[threadIdx.x] = (s == 0.f) ? 0.f : 1.f;
  }
  __syncthreads();

  for (int idx = threadIdx.x; idx < 16 * HD; idx += blockDim.x) {
    int nr = idx >> 7, h = idx & (HD - 1);
    int row = mBase + nr;
    if (row >= M) continue;
    float r  = sigmoidf_(gi[nr][h]      + gh[nr][h]);
    float z  = sigmoidf_(gi[nr][HD + h] + gh[nr][HD + h]);
    float nn = tanhf(gi[nr][2 * HD + h] + r * gh[nr][2 * HD + h]);
    size_t off = (size_t)row * HD + h;
    float h0v = hidden[off];
    hidden[off] = ((1.f - z) * nn + z * h0v) * keep[nr];
  }
}

// ---------------------------------------------------------------------------
// Small helpers
// ---------------------------------------------------------------------------
__global__ void fill_zero(float* __restrict__ p, size_t n) {
  size_t i = (size_t)blockIdx.x * blockDim.x + threadIdx.x;
  if (i < n) p[i] = 0.f;
}

__global__ void cvt_f16(_Float16* __restrict__ dst, const float* __restrict__ src, size_t n) {
  size_t i = (size_t)blockIdx.x * blockDim.x + threadIdx.x;
  if (i < n) dst[i] = (_Float16)src[i];
}

__global__ void init_hidden_kernel(float* __restrict__ hidden, const float* __restrict__ qre,
                                   const int* __restrict__ q_rel, const int* __restrict__ qsub) {
  int q = blockIdx.x, h = threadIdx.x;
  hidden[(size_t)qsub[q] * HD + h] = qre[(size_t)q_rel[q] * HD + h];
}

__global__ void gather_rows16(_Float16* __restrict__ dst, const _Float16* __restrict__ src,
                              const int* __restrict__ idx) {
  int q = blockIdx.x, h = threadIdx.x;
  dst[(size_t)q * HD + h] = src[(size_t)idx[q] * HD + h];
}

// scores[n] = hidden[n] . Wfinal, scattered to out[batch, abs]. One wave/node.
__global__ void score_kernel(const float* __restrict__ hidden, const float* __restrict__ Wfinal,
                             const int* __restrict__ batch_idxs, const int* __restrict__ abs_idxs,
                             float* __restrict__ out, int nNode, int nEnt) {
  int wid = blockIdx.x * (blockDim.x >> 5) + (threadIdx.x >> 5);
  int lane = threadIdx.x & 31;
  if (wid >= nNode) return;
  const float4 h4 = ((const float4*)(hidden + (size_t)wid * HD))[lane];
  const float4 w4 = ((const float4*)Wfinal)[lane];
  float p = h4.x * w4.x + h4.y * w4.y + h4.z * w4.z + h4.w * w4.w;
#pragma unroll
  for (int off = 16; off > 0; off >>= 1) p += __shfl_xor(p, off, 32);
  if (lane == 0) out[(size_t)batch_idxs[wid] * nEnt + abs_idxs[wid]] = p;
}

// ---------------------------------------------------------------------------
// Host orchestration
// ---------------------------------------------------------------------------
extern "C" void kernel_launch(void* const* d_in, const int* in_sizes, int n_in,
                              void* d_out, int out_size, void* d_ws, size_t ws_size,
                              hipStream_t stream) {
  (void)n_in; (void)ws_size;
  const float* rela_embed = (const float*)d_in[0];
  const float* query_rela = (const float*)d_in[1];
  const float* proj_W     = (const float*)d_in[2];
  const float* proj_b     = (const float*)d_in[3];
  const float* Ws         = (const float*)d_in[4];
  const float* Wr         = (const float*)d_in[5];
  const float* WqrW       = (const float*)d_in[6];
  const float* Wqrb       = (const float*)d_in[7];
  const float* walW       = (const float*)d_in[8];
  const float* walB       = (const float*)d_in[9];
  const float* Wh         = (const float*)d_in[10];
  const float* gWih       = (const float*)d_in[11];
  const float* gWhh       = (const float*)d_in[12];
  const float* gbih       = (const float*)d_in[13];
  const float* gbhh       = (const float*)d_in[14];
  const float* Wfinal     = (const float*)d_in[15];
  const int* q_rel      = (const int*)d_in[16];
  const int* batch_idxs = (const int*)d_in[17];
  const int* abs_idxs   = (const int*)d_in[18];
  const int* qsub       = (const int*)d_in[19];
  const int* e_batch    = (const int*)d_in[20];
  const int* e_sub      = (const int*)d_in[21];
  const int* e_rel      = (const int*)d_in[22];
  const int* e_obj      = (const int*)d_in[23];

  const int R      = in_sizes[1] / HD;          // 401
  const int LLM    = in_sizes[0] / R;           // 768
  const int L      = in_sizes[3] / HD;          // 3
  const int nQuery = in_sizes[16];              // 32
  const int nNode  = in_sizes[17];              // 100000
  const int nEdge  = in_sizes[20];              // 500000
  const int nEnt   = out_size / nQuery;         // 50000

  // ---- workspace carve-out (256B-aligned byte slices) ----
  char* base = (char*)d_ws;
  size_t off = 0;
  auto allocB = [&](size_t bytes) {
    void* p = base + off;
    off += (bytes + 255) & ~(size_t)255;
    return p;
  };
  const size_t nh = (size_t)nNode * HD;
  // f32 buffers
  float* hidden   = (float*)allocB(nh * 4);
  float* wsn      = (float*)allocB(nh * 4);
  float* agg      = (float*)allocB(nh * 4);
  float* hlay     = (float*)allocB(nh * 4);
  float* hr_table = (float*)allocB((size_t)R * HD * 4);
  float* wr_rel   = (float*)allocB((size_t)R * HD * 4);
  float* wqr_q    = (float*)allocB((size_t)nQuery * HD * 4);
  // f16 operand buffers
  _Float16* hid16   = (_Float16*)allocB(nh * 2);
  _Float16* a16     = (_Float16*)allocB(nh * 2);          // agg16 then hlay16
  _Float16* rela16  = (_Float16*)allocB((size_t)R * LLM * 2);
  _Float16* hr16    = (_Float16*)allocB((size_t)R * HD * 2);
  _Float16* qhr16   = (_Float16*)allocB((size_t)nQuery * HD * 2);
  _Float16* projW16 = (_Float16*)allocB((size_t)HD * LLM * 2);
  _Float16* Wr16    = (_Float16*)allocB((size_t)HD * HD * 2);
  _Float16* Wq16    = (_Float16*)allocB((size_t)HD * HD * 2);
  _Float16* Ws16    = (_Float16*)allocB((size_t)HD * HD * 2);
  _Float16* Wh16    = (_Float16*)allocB((size_t)HD * HD * 2);
  _Float16* Wih16   = (_Float16*)allocB((size_t)GN * HD * 2);
  _Float16* Whh16   = (_Float16*)allocB((size_t)GN * HD * 2);

  auto cvt = [&](_Float16* dst, const float* src, size_t n) {
    cvt_f16<<<(int)((n + 255) / 256), 256, 0, stream>>>(dst, src, n);
  };

  // ---- layer-invariant conversions ----
  cvt(rela16, rela_embed, (size_t)R * LLM);
  cvt(Wih16, gWih, (size_t)GN * HD);
  cvt(Whh16, gWhh, (size_t)GN * HD);

  // hidden = 0 ; query nodes seeded with query-relation embedding
  fill_zero<<<(int)((nh + 255) / 256), 256, 0, stream>>>(hidden, nh);
  init_hidden_kernel<<<nQuery, HD, 0, stream>>>(hidden, query_rela, q_rel, qsub);

  const dim3 gRel((R + 15) / 16, (HD / 16 + 3) / 4);
  const dim3 gQry((nQuery + 15) / 16, (HD / 16 + 3) / 4);
  const dim3 gNode((nNode + 15) / 16, (HD / 16 + 3) / 4);
  const int gruBlocks = (nNode + 15) / 16;

  for (int i = 0; i < L; ++i) {
    const float* pb    = proj_b + (size_t)i * HD;
    const float* Wqb_i = Wqrb + (size_t)i * HD;
    const float* waW_i = walW + (size_t)i * HD;
    const float* waB_i = walB + i;

    // per-layer weight conversions (tiny, L2-resident afterwards)
    cvt(projW16, proj_W + (size_t)i * HD * LLM, (size_t)HD * LLM);
    cvt(Wr16, Wr + (size_t)i * HD * HD, (size_t)HD * HD);
    cvt(Wq16, WqrW + (size_t)i * HD * HD, (size_t)HD * HD);
    cvt(Ws16, Ws + (size_t)i * HD * HD, (size_t)HD * HD);
    cvt(Wh16, Wh + (size_t)i * HD * HD, (size_t)HD * HD);

    // relation tables
    wmma_gemm_atb<<<gRel, 128, 0, stream>>>(rela16, projW16, pb, hr_table, R, HD, LLM, 0);
    cvt(hr16, hr_table, (size_t)R * HD);
    wmma_gemm_atb<<<gRel, 128, 0, stream>>>(hr16, Wr16, nullptr, wr_rel, R, HD, HD, 0);
    gather_rows16<<<nQuery, HD, 0, stream>>>(qhr16, hr16, q_rel);
    wmma_gemm_atb<<<gQry, 128, 0, stream>>>(qhr16, Wq16, Wqb_i, wqr_q, nQuery, HD, HD, 0);

    // hs@Ws^T hoisted to nodes: wsn = hidden @ Ws^T (E-GEMM -> N-GEMM, 5x fewer FLOPs)
    cvt(hid16, hidden, nh);
    wmma_gemm_atb<<<gNode, 128, 0, stream>>>(hid16, Ws16, nullptr, wsn, nNode, HD, HD, 0);

    // segment-sum of attention-weighted messages
    fill_zero<<<(int)((nh + 255) / 256), 256, 0, stream>>>(agg, nh);
    edge_kernel<<<(nEdge + 7) / 8, 256, 0, stream>>>(e_sub, e_rel, e_batch, e_obj,
                                                     wsn, wr_rel, wqr_q, hidden, hr_table,
                                                     waW_i, waB_i, agg, nEdge);

    // h_layer = relu(agg @ Wh^T)
    cvt(a16, agg, nh);
    wmma_gemm_atb<<<gNode, 128, 0, stream>>>(a16, Wh16, nullptr, hlay, nNode, HD, HD, 1);

    // fused GRU + dead mask, in-place on hidden (h0 == previous hidden; hid16 == h0 in f16)
    cvt(a16, hlay, nh);
    gru_kernel<<<gruBlocks, 256, 0, stream>>>(a16, hid16, hlay, hidden,
                                              Wih16, Whh16, gbih, gbhh, nNode);
  }

  // readout + scatter into [nQuery, nEnt]
  fill_zero<<<(out_size + 255) / 256, 256, 0, stream>>>((float*)d_out, (size_t)out_size);
  score_kernel<<<(nNode + 7) / 8, 256, 0, stream>>>(hidden, Wfinal, batch_idxs, abs_idxs,
                                                    (float*)d_out, nNode, nEnt);
}
